// GlmImageVisionIBQ_7430293422279
// MI455X (gfx1250) — compile-verified, matrix operand order checked by reading, not run on hardware
//
#include <hip/hip_runtime.h>
#include <hip/hip_bf16.h>

typedef __attribute__((ext_vector_type(16))) __bf16 v16bf;
typedef __attribute__((ext_vector_type(8)))  __bf16 v8bf;
typedef __attribute__((ext_vector_type(8)))  float  v8f;

#define NROWS 8192      // B*H*W
#define KDIM  256
#define NEMB  16384
#define NSEG  8
#define SEGN  (NEMB / NSEG)   // 2048
#define CHUNK 32              // codebook rows per LDS buffer
#define BROWS 264             // padded bf16 stride (528 B): conflict-free ds_read_b128
#define BUFB  (CHUNK * BROWS) // bf16 elems per buffer

// ---------------- normalize embeddings -> bf16 [16384,256] ----------------
__global__ __launch_bounds__(256) void k_norm_emb(const float* __restrict__ emb,
                                                  __bf16* __restrict__ En) {
  const int row = (blockIdx.x * 256 + threadIdx.x) >> 5;   // one wave per row
  const int lane = threadIdx.x & 31;
  const float* src = emb + (size_t)row * KDIM;
  float v[8]; float s = 0.f;
#pragma unroll
  for (int t = 0; t < 8; ++t) { v[t] = src[lane + 32 * t]; s += v[t] * v[t]; }
#pragma unroll
  for (int m = 1; m < 32; m <<= 1) s += __shfl_xor(s, m, 32);
  const float inv = 1.0f / fmaxf(sqrtf(s), 1e-12f);
  __bf16* dst = En + (size_t)row * KDIM;
#pragma unroll
  for (int t = 0; t < 8; ++t) dst[lane + 32 * t] = (__bf16)(v[t] * inv);
}

// ---------------- normalize z (NCHW -> rows) -> bf16 [8192,256] -----------
__global__ __launch_bounds__(256) void k_norm_z(const float* __restrict__ z,
                                                __bf16* __restrict__ Zn) {
  __shared__ float lds[32 * 258];                 // [w][c], padded stride
  const int b = blockIdx.x >> 5, h = blockIdx.x & 31;
  const int tid = threadIdx.x;
  const int w = tid & 31, c0 = tid >> 5;
  const float* zb = z + (((size_t)b * 256) * 32 + h) * 32;   // + c*1024 + w
#pragma unroll
  for (int i = 0; i < 32; ++i) {
    const int c = c0 + i * 8;
    lds[w * 258 + c] = zb[(size_t)c * 1024 + w];
  }
  __syncthreads();
  const int wave = tid >> 5, lane = tid & 31;
  for (int j = 0; j < 4; ++j) {
    const int ww = wave * 4 + j;
    const float* row = &lds[ww * 258];
    float v[8]; float s = 0.f;
#pragma unroll
    for (int t = 0; t < 8; ++t) { v[t] = row[lane + 32 * t]; s += v[t] * v[t]; }
#pragma unroll
    for (int m = 1; m < 32; m <<= 1) s += __shfl_xor(s, m, 32);
    const float inv = 1.0f / fmaxf(sqrtf(s), 1e-12f);
    const int n = b * 1024 + h * 32 + ww;
    __bf16* dst = Zn + (size_t)n * KDIM;
#pragma unroll
    for (int t = 0; t < 8; ++t) dst[lane + 32 * t] = (__bf16)(v[t] * inv);
  }
}

// ------------- fused GEMM (bf16 WMMA) + per-row running argmax ------------
// 3-deep async (GLOBAL_LOAD_ASYNC_TO_LDS_B128) pipeline: issue chunk i+1,
// compute chunk i; s_wait_asynccnt 4 leaves only newest 4 ops in flight.
__global__ __launch_bounds__(256) void k_gemm_argmax(const __bf16* __restrict__ Zn,
                                                     const __bf16* __restrict__ En,
                                                     float* __restrict__ part_val,
                                                     int* __restrict__ part_idx) {
  __shared__ __align__(16) __bf16 Bs[3 * BUFB];   // 3 x 16.5 KB padded buffers
  const int tid  = threadIdx.x;
  const int lane = tid & 31;
  const int wave = tid >> 5;
  const int half = lane >> 4;
  const int lmod = lane & 15;
  const int mbase = blockIdx.x * 128 + wave * 16;  // 16 rows per wave
  const int seg = blockIdx.y;
  const int n_begin = seg * SEGN;
  const int n_end   = n_begin + SEGN;

  // A fragments for this wave's 16 rows, full K=256 (8 k-steps), resident
  v16bf a[8];
  {
    const __bf16* arow = Zn + (size_t)(mbase + lmod) * KDIM;
#pragma unroll
    for (int kk = 0; kk < 8; ++kk) {
      v8bf lo = *(const v8bf*)(arow + kk * 32 + half * 8);
      v8bf hi = *(const v8bf*)(arow + kk * 32 + half * 8 + 16);
      a[kk] = __builtin_shufflevector(lo, hi, 0,1,2,3,4,5,6,7,8,9,10,11,12,13,14,15);
    }
  }

  // per-thread async transfer offsets: chunk = 32 rows x 512 B = 1024 b128 ops
  const unsigned ldsBase = (unsigned)(uintptr_t)(&Bs[0]);   // low 32b = LDS offset
  unsigned dstOff[4], srcOff[4];
#pragma unroll
  for (int i = 0; i < 4; ++i) {
    const int lin = tid + 256 * i;                 // 0..1023
    dstOff[i] = (unsigned)((lin >> 5) * 528 + (lin & 31) * 16);  // padded LDS
    srcOff[i] = (unsigned)(lin * 16);              // byte offset in 16 KB chunk
  }

  auto issue_chunk = [&](unsigned bufByteBase, int n0) {
    const char* sb = (const char*)(En + (size_t)n0 * KDIM);
#pragma unroll
    for (int i = 0; i < 4; ++i) {
      asm volatile("global_load_async_to_lds_b128 %0, %1, %2"
                   :: "v"(bufByteBase + dstOff[i]), "v"(srcOff[i]), "s"(sb)
                   : "memory");
    }
  };

  float best[8]; int bidx[8];
#pragma unroll
  for (int r = 0; r < 8; ++r) { best[r] = -__builtin_inff(); bidx[r] = 0; }

  issue_chunk(ldsBase, n_begin);                   // prime buffer 0
  int p = 0;
  for (int n0 = n_begin; n0 < n_end; n0 += CHUNK) {
    if (n0 + CHUNK < n_end) {
      issue_chunk(ldsBase + (unsigned)(((p + 1) % 3) * (BUFB * 2)), n0 + CHUNK);
      asm volatile("s_wait_asynccnt 0x4" ::: "memory");  // current chunk landed
    } else {
      asm volatile("s_wait_asynccnt 0x0" ::: "memory");
    }
    __syncthreads();

    const __bf16* bufp = &Bs[p * BUFB];
#pragma unroll
    for (int t = 0; t < 2; ++t) {
      const __bf16* brow = bufp + (t * 16 + lmod) * BROWS;
      v8f c = {0.f, 0.f, 0.f, 0.f, 0.f, 0.f, 0.f, 0.f};
#pragma unroll
      for (int kk = 0; kk < 8; ++kk) {
        v8bf lo = *(const v8bf*)(brow + kk * 32 + half * 8);
        v8bf hi = *(const v8bf*)(brow + kk * 32 + half * 8 + 16);
        v16bf bf = __builtin_shufflevector(lo, hi, 0,1,2,3,4,5,6,7,8,9,10,11,12,13,14,15);
        c = __builtin_amdgcn_wmma_f32_16x16x32_bf16(false, a[kk], false, bf,
                                                    (short)0, c, false, false);
      }
      const int n = n0 + t * 16 + lmod;            // this lane's column
#pragma unroll
      for (int r = 0; r < 8; ++r)
        if (c[r] > best[r]) { best[r] = c[r]; bidx[r] = n; }  // strict > keeps first idx
    }

    // Scheduling hint: keep DS reads one stage ahead of the WMMA chain so the
    // wait inserter can use partial s_wait_dscnt instead of dscnt==0 stalls.
    // 0x100 = DS-read group, 0x008 = MFMA/WMMA group.
    __builtin_amdgcn_sched_group_barrier(0x100, 4, 0);    // 2 pairs up front
#pragma unroll
    for (int i = 0; i < 14; ++i) {
      __builtin_amdgcn_sched_group_barrier(0x008, 1, 0);  // 1 wmma
      __builtin_amdgcn_sched_group_barrier(0x100, 2, 0);  // next pair
    }
    __builtin_amdgcn_sched_group_barrier(0x008, 2, 0);    // drain last 2 wmma

    p = (p + 1) % 3;
  }

  // reduce across the 16 lanes of each half (rows r / r+8)
#pragma unroll
  for (int m = 1; m < 16; m <<= 1) {
#pragma unroll
    for (int r = 0; r < 8; ++r) {
      const float ov = __shfl_xor(best[r], m, 32);
      const int   oi = __shfl_xor(bidx[r], m, 32);
      if (ov > best[r] || (ov == best[r] && oi < bidx[r])) { best[r] = ov; bidx[r] = oi; }
    }
  }
  if (lmod == 0) {                      // lane 0 -> rows 0..7, lane 16 -> rows 8..15
    const int rb = mbase + half * 8;
#pragma unroll
    for (int r = 0; r < 8; ++r) {
      part_val[(size_t)(rb + r) * NSEG + seg] = best[r];
      part_idx[(size_t)(rb + r) * NSEG + seg] = bidx[r];
    }
  }
}

// ---------------- combine segment partials -> final indices ----------------
__global__ __launch_bounds__(256) void k_reduce(const float* __restrict__ pv,
                                                const int* __restrict__ pi,
                                                int* __restrict__ idxs,
                                                float* __restrict__ idx_f) {
  const int n = blockIdx.x * 256 + threadIdx.x;
  float bv = pv[(size_t)n * NSEG];
  int   bi = pi[(size_t)n * NSEG];
  for (int s = 1; s < NSEG; ++s) {
    const float v = pv[(size_t)n * NSEG + s];
    const int   i = pi[(size_t)n * NSEG + s];
    if (v > bv || (v == bv && i < bi)) { bv = v; bi = i; }
  }
  idxs[n] = bi;
  idx_f[n] = (float)bi;
}

// -------- gather normalized fp32 embedding rows, scatter to NCHW ----------
__global__ __launch_bounds__(256) void k_gather(const float* __restrict__ emb,
                                                const int* __restrict__ idxs,
                                                float* __restrict__ zq) {
  __shared__ float lds[32 * 258];
  const int b = blockIdx.x >> 5, h = blockIdx.x & 31;
  const int tid = threadIdx.x;
  const int wave = tid >> 5, lane = tid & 31;
  for (int j = 0; j < 4; ++j) {
    const int ww = wave * 4 + j;
    const int n = b * 1024 + h * 32 + ww;
    const float* row = emb + (size_t)idxs[n] * KDIM;
    float v[8]; float s = 0.f;
#pragma unroll
    for (int t = 0; t < 8; ++t) { v[t] = row[lane + 32 * t]; s += v[t] * v[t]; }
#pragma unroll
    for (int m = 1; m < 32; m <<= 1) s += __shfl_xor(s, m, 32);
    const float inv = 1.0f / fmaxf(sqrtf(s), 1e-12f);
#pragma unroll
    for (int t = 0; t < 8; ++t) lds[ww * 258 + lane + 32 * t] = v[t] * inv;
  }
  __syncthreads();
  const int w = tid & 31, c0 = tid >> 5;
  float* zb = zq + (((size_t)b * 256) * 32 + h) * 32;
#pragma unroll
  for (int i = 0; i < 32; ++i) {
    const int c = c0 + i * 8;
    zb[(size_t)c * 1024 + w] = lds[w * 258 + c];
  }
}

extern "C" void kernel_launch(void* const* d_in, const int* in_sizes, int n_in,
                              void* d_out, int out_size, void* d_ws, size_t ws_size,
                              hipStream_t stream) {
  const float* z   = (const float*)d_in[0];   // [8,256,32,32] f32
  const float* emb = (const float*)d_in[1];   // [16384,256]   f32
  float* out = (float*)d_out;                 // z_q (2097152 f32) ++ indices (8192)

  char* ws = (char*)d_ws;
  __bf16* En      = (__bf16*)(ws);                         //  8,388,608 B
  __bf16* Zn      = (__bf16*)(ws + 8388608);               //  4,194,304 B
  float*  part_val = (float*)(ws + 12582912);              //    262,144 B
  int*    part_idx = (int*)  (ws + 12845056);              //    262,144 B
  int*    idxs     = (int*)  (ws + 13107200);              //     32,768 B

  float* zq    = out;
  float* idx_f = out + 2097152;

  k_norm_emb<<<NEMB / 8, 256, 0, stream>>>(emb, En);            // 8 waves/block
  k_norm_z<<<256, 256, 0, stream>>>(z, Zn);                     // one (b,h) per block
  k_gemm_argmax<<<dim3(NROWS / 128, NSEG), 256, 0, stream>>>(Zn, En, part_val, part_idx);
  k_reduce<<<NROWS / 256, 256, 0, stream>>>(part_val, part_idx, idxs, idx_f);
  k_gather<<<256, 256, 0, stream>>>(emb, idxs, zq);
}
